// LinearBlock_62783831933559
// MI455X (gfx1250) — matrix operand order repaired and checked by reading, not compiled, not run on hardware
//
#include <hip/hip_runtime.h>
#include <hip/hip_bf16.h>

// ---------------------------------------------------------------------------
// Problem constants (from reference)
// ---------------------------------------------------------------------------
#define DIMC   768
#define HEADSC 12
#define EC     64
#define HIDC   3072
#define NQKV   2304          // 3*DIMC
#define BATCHC 8
#define FRAMESC 16
#define NTOKC  196           // 14*14
#define MTOK   25088         // BATCH*FRAMES*NTOK
#define MSEQ   3136          // FRAMES*NTOK (joint attention length)
#define NSEG   7             // m-reduction segments (448 = 14*32 each)
#define SEGM   448

typedef __bf16 v16bf __attribute__((ext_vector_type(16)));
typedef float  v8f   __attribute__((ext_vector_type(8)));

union FragU { uint4 q[2]; v16bf v; };

__device__ __forceinline__ unsigned short f2bf(float f) {
    unsigned int u = __float_as_uint(f);
    u += 0x7FFFu + ((u >> 16) & 1u);          // round-to-nearest-even
    return (unsigned short)(u >> 16);
}
__device__ __forceinline__ float bf2f(unsigned short h) {
    return __uint_as_float(((unsigned int)h) << 16);
}

// ---------------------------------------------------------------------------
// f32 -> bf16 elementwise (weight conversion)
// ---------------------------------------------------------------------------
__global__ void f32_to_bf16_kernel(const float* __restrict__ in,
                                   unsigned short* __restrict__ out, int n) {
    int i = blockIdx.x * 256 + threadIdx.x;
    if (i < n) out[i] = f2bf(in[i]);
}

// ---------------------------------------------------------------------------
// LayerNorm: wave-per-row (wave32), 8 rows per block, fp32 in -> bf16 out
// ---------------------------------------------------------------------------
__global__ __launch_bounds__(256) void layernorm_bf16_kernel(
    const float* __restrict__ x, const float* __restrict__ w,
    const float* __restrict__ b, unsigned short* __restrict__ out) {
    int row  = blockIdx.x * 8 + (threadIdx.x >> 5);
    int lane = threadIdx.x & 31;
    const float* xr = x + (size_t)row * DIMC;
    float v[24];
    float s = 0.f;
    #pragma unroll
    for (int i = 0; i < 24; i++) { v[i] = xr[lane + i * 32]; s += v[i]; }
    #pragma unroll
    for (int m = 16; m; m >>= 1) s += __shfl_xor(s, m, 32);
    float mu = s * (1.f / DIMC);
    float q = 0.f;
    #pragma unroll
    for (int i = 0; i < 24; i++) { float d = v[i] - mu; q += d * d; }
    #pragma unroll
    for (int m = 16; m; m >>= 1) q += __shfl_xor(q, m, 32);
    float rs = rsqrtf(q * (1.f / DIMC) + 1e-5f);
    #pragma unroll
    for (int i = 0; i < 24; i++) {
        int c = lane + i * 32;
        out[(size_t)row * DIMC + c] = f2bf((v[i] - mu) * rs * w[c] + b[c]);
    }
}

// ---------------------------------------------------------------------------
// Tiled bf16 WMMA GEMM:  out[m,n] = sum_k A[m,k] * W[n,k]  (+ epilogue)
// Block tile 128x256, 8 waves (2x4), wave tile 64x64 (4x4 wmma 16x16x32).
// ---------------------------------------------------------------------------
#define EPI_QKV  0   // q/k cols: relu(v)+0.125, v cols plain -> bf16
#define EPI_PROJ 1   // + bias + residual(add_src) -> fp32
#define EPI_FC1  2   // + bias -> bf16
#define EPI_FC2  3   // + bias + out_f32 (accumulate) -> fp32

template <int EPI>
__global__ __launch_bounds__(256) void gemm_bf16_wmma_kernel(
    const unsigned short* __restrict__ A,   // M x K, row-major bf16
    const unsigned short* __restrict__ W,   // N x K, row-major bf16
    int M, int N, int K,
    const float* __restrict__ bias,
    const float* __restrict__ add_src,
    float* __restrict__ out_f32,
    unsigned short* __restrict__ out_bf16) {
    __shared__ alignas(16) unsigned short sA[128 * 32];
    __shared__ alignas(16) unsigned short sB[256 * 32];

    const int tid  = threadIdx.x;
    const int lane = tid & 31;
    const int wave = tid >> 5;
    const int wm   = wave >> 2;      // 0..1 : wave row (64 rows each)
    const int wn   = wave & 3;       // 0..3 : wave col (64 cols each)
    const int bm   = blockIdx.y * 128;
    const int bn   = blockIdx.x * 256;
    const int l15  = lane & 15;
    const int lhi  = lane >> 4;

    v8f acc[4][4];
    #pragma unroll
    for (int i = 0; i < 4; i++)
        #pragma unroll
        for (int j = 0; j < 4; j++)
            #pragma unroll
            for (int e = 0; e < 8; e++) acc[i][j][e] = 0.f;

    // per-thread staging coordinates (branch-free)
    const int rA = tid >> 2;               // 0..63   (A rows, 2 passes)
    const int rB = tid >> 2;               // 0..63   (B rows, 4 passes)
    const int c8 = (tid & 3) << 3;         // 0,8,16,24

    for (int kt = 0; kt < K; kt += 32) {
        __syncthreads();
        // A tile 128x32: 512 x 16B chunks, 2 per thread
        #pragma unroll
        for (int it = 0; it < 2; it++) {
            int r = rA + it * 64;
            *(uint4*)&sA[r * 32 + c8] =
                *(const uint4*)&A[(size_t)(bm + r) * K + kt + c8];
        }
        // B tile 256x32: 1024 x 16B chunks, 4 per thread
        #pragma unroll
        for (int it = 0; it < 4; it++) {
            int r = rB + it * 64;
            *(uint4*)&sB[r * 32 + c8] =
                *(const uint4*)&W[(size_t)(bn + r) * K + kt + c8];
        }
        // pull the tile after next toward L2 (global_prefetch_b8)
        if (kt + 64 < K) {
            __builtin_prefetch(&A[(size_t)(bm + (tid >> 1)) * K + kt + 64], 0, 3);
            __builtin_prefetch(&W[(size_t)(bn + tid) * K + kt + 64], 0, 3);
        }
        __syncthreads();

        // A fragments: 16x32 bf16, lane<16 -> K 0..7 & 16..23, lane>=16 -> +8
        FragU afr[4], bfr[4];
        const int kb = lhi * 8;
        #pragma unroll
        for (int tm = 0; tm < 4; tm++) {
            int rr = wm * 64 + tm * 16 + l15;
            afr[tm].q[0] = *(const uint4*)&sA[rr * 32 + kb];
            afr[tm].q[1] = *(const uint4*)&sA[rr * 32 + kb + 16];
        }
        // B fragments: 32x16 bf16, lane<16 -> K 0..7 & 16..23, lane>=16 -> +8
        const int kb2 = lhi * 8;
        #pragma unroll
        for (int tn = 0; tn < 4; tn++) {
            int cc = wn * 64 + tn * 16 + l15;
            bfr[tn].q[0] = *(const uint4*)&sB[cc * 32 + kb2];
            bfr[tn].q[1] = *(const uint4*)&sB[cc * 32 + kb2 + 16];
        }
        #pragma unroll
        for (int tm = 0; tm < 4; tm++)
            #pragma unroll
            for (int tn = 0; tn < 4; tn++)
                acc[tm][tn] = __builtin_amdgcn_wmma_f32_16x16x32_bf16(
                    false, afr[tm].v, false, bfr[tn].v,
                    (short)0, acc[tm][tn], false, false);
    }

    // epilogue: VGPR r -> M = r + 8*(lane>=16), N = lane&15
    #pragma unroll
    for (int tm = 0; tm < 4; tm++)
        #pragma unroll
        for (int tn = 0; tn < 4; tn++)
            #pragma unroll
            for (int r = 0; r < 8; r++) {
                int gm = bm + wm * 64 + tm * 16 + r + (lhi << 3);
                int gn = bn + wn * 64 + tn * 16 + l15;
                float v = acc[tm][tn][r];
                size_t o = (size_t)gm * N + gn;
                if (EPI == EPI_QKV) {
                    if (gn < 2 * DIMC) v = fmaxf(v, 0.f) + 0.125f; // relu+e^-0.5
                    out_bf16[o] = f2bf(v);
                } else if (EPI == EPI_PROJ) {
                    out_f32[o] = v + bias[gn] + add_src[o];
                } else if (EPI == EPI_FC1) {
                    out_bf16[o] = f2bf(v + bias[gn]);
                } else { // EPI_FC2
                    out_f32[o] = out_f32[o] + v + bias[gn];
                }
            }
}

// ---------------------------------------------------------------------------
// ksum partials: ksp[seg][b,h,e] = sum_{m in seg} k_[b,m,h,e]
// ---------------------------------------------------------------------------
__global__ void ksum_part_kernel(const unsigned short* __restrict__ qkv,
                                 float* __restrict__ ksp) {
    int bh = blockIdx.x;
    int seg = blockIdx.y;
    int b = bh / HEADSC, h = bh % HEADSC;
    int e = threadIdx.x;  // 0..63
    const unsigned short* base =
        qkv + (size_t)b * MSEQ * NQKV + (size_t)seg * SEGM * NQKV + DIMC + h * EC + e;
    float s = 0.f;
    for (int m = 0; m < SEGM; m++) s += bf2f(base[(size_t)m * NQKV]);
    ksp[(size_t)seg * (BATCHC * HEADSC * EC) + bh * EC + e] = s;
}

// ---------------------------------------------------------------------------
// kv partials over m-segments: kvp[seg][b,h,e,d] (LDS-staged VALU, 64x64xSEGM)
// ---------------------------------------------------------------------------
__global__ __launch_bounds__(256) void kv_part_kernel(
    const unsigned short* __restrict__ qkv, float* __restrict__ kvp) {
    __shared__ alignas(16) unsigned short sk[32 * 64];
    __shared__ alignas(16) unsigned short sv[32 * 64];
    int bh = blockIdx.x;
    int seg = blockIdx.y;
    int b = bh / HEADSC, h = bh % HEADSC;
    int tid = threadIdx.x;
    int e0 = ((tid >> 4) & 15) * 4;
    int d0 = (tid & 15) * 4;
    float acc[4][4] = {};
    size_t mbase = (size_t)b * MSEQ + (size_t)seg * SEGM;
    const unsigned short* kb = qkv + mbase * NQKV + DIMC + h * EC;
    const unsigned short* vb = qkv + mbase * NQKV + 2 * DIMC + h * EC;
    for (int m0 = 0; m0 < SEGM; m0 += 32) {
        __syncthreads();
        int r = tid >> 3, seg8 = (tid & 7) << 3;
        *(uint4*)&sk[r * 64 + seg8] = *(const uint4*)&kb[(size_t)(m0 + r) * NQKV + seg8];
        *(uint4*)&sv[r * 64 + seg8] = *(const uint4*)&vb[(size_t)(m0 + r) * NQKV + seg8];
        __syncthreads();
        for (int m = 0; m < 32; m++) {
            float kr[4], vr[4];
            #pragma unroll
            for (int i = 0; i < 4; i++) kr[i] = bf2f(sk[m * 64 + e0 + i]);
            #pragma unroll
            for (int j = 0; j < 4; j++) vr[j] = bf2f(sv[m * 64 + d0 + j]);
            #pragma unroll
            for (int i = 0; i < 4; i++)
                #pragma unroll
                for (int j = 0; j < 4; j++) acc[i][j] += kr[i] * vr[j];
        }
    }
    float* dst = kvp + (size_t)seg * (BATCHC * HEADSC * EC * EC) + (size_t)bh * EC * EC;
    #pragma unroll
    for (int i = 0; i < 4; i++)
        #pragma unroll
        for (int j = 0; j < 4; j++)
            dst[(e0 + i) * EC + d0 + j] = acc[i][j];
}

// ---------------------------------------------------------------------------
// dst[i] = sum_{s<segs} src[s*n + i]   (deterministic 2nd-stage reduction)
// ---------------------------------------------------------------------------
__global__ void reduce_seg_kernel(const float* __restrict__ src,
                                  float* __restrict__ dst, int n, int segs) {
    int i = blockIdx.x * 256 + threadIdx.x;
    if (i >= n) return;
    float s = 0.f;
    for (int k = 0; k < segs; k++) s += src[(size_t)k * n + i];
    dst[i] = s;
}

// ---------------------------------------------------------------------------
// out[b,m,h,d] = (q_[b,m,h,:] . kv[b,h,:,d]) / (q_ . ksum + eps)  -> bf16
// ---------------------------------------------------------------------------
__global__ __launch_bounds__(256) void attn_out_kernel(
    const unsigned short* __restrict__ qkv, const float* __restrict__ kv,
    const float* __restrict__ ksum, unsigned short* __restrict__ out) {
    __shared__ alignas(16) unsigned short sq[64 * 64];
    __shared__ float skv[64 * 64];
    __shared__ float sks[64];
    __shared__ float sz[64];
    int bh = blockIdx.x;
    int b = bh / HEADSC, h = bh % HEADSC;
    int m0 = blockIdx.y * 64;
    int tid = threadIdx.x;
    const unsigned short* qb = qkv + (size_t)b * MSEQ * NQKV + h * EC;
    for (int idx = tid; idx < 512; idx += 256) {
        int r = idx >> 3, seg = (idx & 7) << 3;
        *(uint4*)&sq[r * 64 + seg] = *(const uint4*)&qb[(size_t)(m0 + r) * NQKV + seg];
    }
    for (int idx = tid; idx < 4096; idx += 256)
        skv[idx] = kv[(size_t)bh * 4096 + idx];
    if (tid < 64) sks[tid] = ksum[bh * EC + tid];
    __syncthreads();
    if (tid < 64) {
        float dot = 0.f;
        for (int e = 0; e < 64; e++) dot += bf2f(sq[tid * 64 + e]) * sks[e];
        sz[tid] = 1.f / (dot + 1e-6f);
    }
    __syncthreads();
    for (int idx = tid; idx < 4096; idx += 256) {
        int ml = idx >> 6, d = idx & 63;
        float s = 0.f;
        for (int e = 0; e < 64; e++)
            s += bf2f(sq[ml * 64 + e]) * skv[e * 64 + d];
        out[((size_t)(b * MSEQ + m0 + ml)) * DIMC + h * EC + d] = f2bf(s * sz[ml]);
    }
}

// ---------------------------------------------------------------------------
// Depthwise 3x3 conv (14x14 grid, zero pad) + bias + exact GELU, bf16->bf16
// ---------------------------------------------------------------------------
__global__ __launch_bounds__(256) void dwconv_gelu_kernel(
    const unsigned short* __restrict__ hdn, const float* __restrict__ w,
    const float* __restrict__ bias, unsigned short* __restrict__ out) {
    __shared__ alignas(16) unsigned short sh[NTOKC * 64];
    __shared__ float sw[64 * 9];
    __shared__ float sb[64];
    int bn = blockIdx.y;            // 0..127 frame index
    int cb = blockIdx.x * 64;       // channel base
    int tid = threadIdx.x;
    for (int idx = tid; idx < NTOKC * 8; idx += 256) {
        int n = idx >> 3, seg = (idx & 7) << 3;
        *(uint4*)&sh[n * 64 + seg] =
            *(const uint4*)&hdn[((size_t)(bn * NTOKC + n)) * HIDC + cb + seg];
    }
    for (int idx = tid; idx < 576; idx += 256)
        sw[idx] = w[(size_t)(cb + idx / 9) * 9 + idx % 9];
    if (tid < 64) sb[tid] = bias[cb + tid];
    __syncthreads();
    for (int idx = tid; idx < NTOKC * 64; idx += 256) {
        int c = idx & 63, n = idx >> 6;
        int y = n / 14, x = n - y * 14;
        float a = sb[c];
        #pragma unroll
        for (int ky = -1; ky <= 1; ky++) {
            int yy = y + ky;
            if (yy < 0 || yy > 13) continue;
            #pragma unroll
            for (int kx = -1; kx <= 1; kx++) {
                int xx = x + kx;
                if (xx < 0 || xx > 13) continue;
                a += sw[c * 9 + (ky + 1) * 3 + (kx + 1)] *
                     bf2f(sh[(yy * 14 + xx) * 64 + c]);
            }
        }
        float g = 0.5f * a * (1.f + erff(a * 0.70710678118654752f));
        out[((size_t)(bn * NTOKC + n)) * HIDC + cb + c] = f2bf(g);
    }
}

// ---------------------------------------------------------------------------
// Host-side orchestration
// ---------------------------------------------------------------------------
extern "C" void kernel_launch(void* const* d_in, const int* in_sizes, int n_in,
                              void* d_out, int out_size, void* d_ws, size_t ws_size,
                              hipStream_t stream) {
    const float* x      = (const float*)d_in[0];
    const float* ln1_w  = (const float*)d_in[1];
    const float* ln1_b  = (const float*)d_in[2];
    const float* qkv_w  = (const float*)d_in[3];
    const float* proj_w = (const float*)d_in[4];
    const float* proj_b = (const float*)d_in[5];
    const float* ln2_w  = (const float*)d_in[6];
    const float* ln2_b  = (const float*)d_in[7];
    const float* fc1_w  = (const float*)d_in[8];
    const float* fc1_b  = (const float*)d_in[9];
    const float* dw_w   = (const float*)d_in[10];
    const float* dw_b   = (const float*)d_in[11];
    const float* fc2_w  = (const float*)d_in[12];
    const float* fc2_b  = (const float*)d_in[13];
    float* out = (float*)d_out;

    // workspace carve-out (all offsets 256B aligned)
    char* p = (char*)d_ws;
    auto carve = [&](size_t bytes) {
        void* r = (void*)p;
        p += (bytes + 255) & ~(size_t)255;
        return r;
    };
    unsigned short* B1  = (unsigned short*)carve((size_t)MTOK * HIDC * 2); // qkv, then hdn
    unsigned short* B2  = (unsigned short*)carve((size_t)MTOK * DIMC * 2); // xn / attn / xn2
    unsigned short* B4  = (unsigned short*)carve((size_t)MTOK * HIDC * 2); // hdn post conv
    float*          KV  = (float*)carve((size_t)BATCHC * HEADSC * EC * EC * 4);
    float*          KS  = (float*)carve((size_t)BATCHC * HEADSC * EC * 4);
    float*          KVP = (float*)carve((size_t)NSEG * BATCHC * HEADSC * EC * EC * 4);
    float*          KSP = (float*)carve((size_t)NSEG * BATCHC * HEADSC * EC * 4);
    unsigned short* Wq  = (unsigned short*)carve((size_t)NQKV * DIMC * 2);
    unsigned short* Wp  = (unsigned short*)carve((size_t)DIMC * DIMC * 2);
    unsigned short* W1  = (unsigned short*)carve((size_t)HIDC * DIMC * 2);
    unsigned short* W2  = (unsigned short*)carve((size_t)DIMC * HIDC * 2);

    auto cvt = [&](const float* src, unsigned short* dst, int n) {
        f32_to_bf16_kernel<<<(n + 255) / 256, 256, 0, stream>>>(src, dst, n);
    };
    cvt(qkv_w,  Wq, NQKV * DIMC);
    cvt(proj_w, Wp, DIMC * DIMC);
    cvt(fc1_w,  W1, HIDC * DIMC);
    cvt(fc2_w,  W2, DIMC * HIDC);

    // 1) xn = LN1(x) -> bf16
    layernorm_bf16_kernel<<<MTOK / 8, 256, 0, stream>>>(x, ln1_w, ln1_b, B2);
    // 2) qkv = xn @ qkv_w^T ; q/k get relu+scale in epilogue
    gemm_bf16_wmma_kernel<EPI_QKV><<<dim3(NQKV / 256, MTOK / 128), 256, 0, stream>>>(
        B2, Wq, MTOK, NQKV, DIMC, nullptr, nullptr, nullptr, B1);
    // 3) linear attention reductions (segmented, deterministic)
    ksum_part_kernel<<<dim3(BATCHC * HEADSC, NSEG), 64, 0, stream>>>(B1, KSP);
    kv_part_kernel<<<dim3(BATCHC * HEADSC, NSEG), 256, 0, stream>>>(B1, KVP);
    reduce_seg_kernel<<<(BATCHC * HEADSC * EC * EC + 255) / 256, 256, 0, stream>>>(
        KVP, KV, BATCHC * HEADSC * EC * EC, NSEG);
    reduce_seg_kernel<<<(BATCHC * HEADSC * EC + 255) / 256, 256, 0, stream>>>(
        KSP, KS, BATCHC * HEADSC * EC, NSEG);
    attn_out_kernel<<<dim3(BATCHC * HEADSC, MSEQ / 64), 256, 0, stream>>>(B1, KV, KS, B2);
    // 4) x2 = x + attn @ proj_w^T + proj_b  -> d_out (fp32)
    gemm_bf16_wmma_kernel<EPI_PROJ><<<dim3(DIMC / 256, MTOK / 128), 256, 0, stream>>>(
        B2, Wp, MTOK, DIMC, DIMC, proj_b, x, out, nullptr);
    // 5) xn2 = LN2(x2) -> bf16
    layernorm_bf16_kernel<<<MTOK / 8, 256, 0, stream>>>(out, ln2_w, ln2_b, B2);
    // 6) hdn = xn2 @ fc1_w^T + fc1_b -> bf16
    gemm_bf16_wmma_kernel<EPI_FC1><<<dim3(HIDC / 256, MTOK / 128), 256, 0, stream>>>(
        B2, W1, MTOK, HIDC, DIMC, fc1_b, nullptr, nullptr, B1);
    // 7) depthwise conv + bias + GELU -> bf16
    dwconv_gelu_kernel<<<dim3(HIDC / 64, BATCHC * FRAMESC), 256, 0, stream>>>(
        B1, dw_w, dw_b, B4);
    // 8) d_out += hdn2 @ fc2_w^T + fc2_b
    gemm_bf16_wmma_kernel<EPI_FC2><<<dim3(DIMC / 256, MTOK / 128), 256, 0, stream>>>(
        B4, W2, MTOK, DIMC, HIDC, fc2_b, nullptr, out, nullptr);
}
